// DeepfakeGNN_18511309045924
// MI455X (gfx1250) — matrix operand order repaired and checked by reading, not compiled, not run on hardware
//
#include <hip/hip_runtime.h>

// Problem constants (match reference)
#define NN   20000
#define EE   320000
#define DIN  512
#define DH   256
#define GG   128

typedef __bf16 bf16_t;
typedef __attribute__((ext_vector_type(16))) __bf16 v16bf;
typedef __attribute__((ext_vector_type(8)))  __bf16 v8bf;
typedef __attribute__((ext_vector_type(8)))  float  v8f;

__device__ __forceinline__ bf16_t to_bf16(float f) { return (bf16_t)f; }

// ---------------------------------------------------------------------------
// Degree / normalization
// ---------------------------------------------------------------------------
__global__ void k_init_deg(float* __restrict__ deg, int n) {
    int i = blockIdx.x * blockDim.x + threadIdx.x;
    for (; i < n; i += gridDim.x * blockDim.x) deg[i] = 1.0f;   // self loop
}

__global__ void k_deg_edges(const int* __restrict__ dst, float* __restrict__ deg, int e) {
    int i = blockIdx.x * blockDim.x + threadIdx.x;
    for (; i < e; i += gridDim.x * blockDim.x) atomicAdd(&deg[dst[i]], 1.0f);
}

__global__ void k_rsqrt_inplace(float* __restrict__ deg, int n) {
    int i = blockIdx.x * blockDim.x + threadIdx.x;
    for (; i < n; i += gridDim.x * blockDim.x) {
        float d = deg[i];
        deg[i] = (d > 0.0f) ? __frsqrt_rn(d) : 0.0f;
    }
}

// ---------------------------------------------------------------------------
// Precision conversion
// ---------------------------------------------------------------------------
__global__ void k_f32_to_bf16(const float* __restrict__ in, bf16_t* __restrict__ out, size_t n) {
    size_t i = (size_t)blockIdx.x * blockDim.x + threadIdx.x;
    size_t stride = (size_t)gridDim.x * blockDim.x;
    for (; i < n; i += stride) out[i] = to_bf16(in[i]);
}

// W[K][N] (f32, row major) -> WT[N][K] (bf16)  : makes B fragments contiguous
__global__ void k_transpose_bf16(const float* __restrict__ W, bf16_t* __restrict__ WT,
                                 int K, int Nn) {
    int i = blockIdx.x * blockDim.x + threadIdx.x;
    int total = K * Nn;
    for (; i < total; i += gridDim.x * blockDim.x) {
        int k = i / Nn, n = i % Nn;
        WT[(size_t)n * K + k] = to_bf16(W[i]);
    }
}

// ---------------------------------------------------------------------------
// WMMA GEMM:  C[M x 256] = A[M x K] * B[K x 256],  BT is B transposed [256 x K]
// block = 128 threads (4 waves); each wave -> 32(M) x 64(N) strip (2 M-tiles
// sharing every B fragment -> 2x arithmetic intensity, back-to-back WMMAs);
// grid = M/32
// ---------------------------------------------------------------------------
template<int KDIM>
__global__ __launch_bounds__(128)
void k_wmma_gemm(const bf16_t* __restrict__ A, const bf16_t* __restrict__ BT,
                 float* __restrict__ C, int M) {
    const int wave  = threadIdx.x >> 5;
    const int lane  = threadIdx.x & 31;
    const int row   = lane & 15;        // A row within tile / D col select
    const int ksel  = lane >> 4;        // which K half-group this lane holds
    const int mbase = blockIdx.x * 32;
    const int nwave = wave * 64;

    const bf16_t* __restrict__ arow0 = A + (size_t)(mbase + row) * KDIM;
    const bf16_t* __restrict__ arow1 = A + (size_t)(mbase + 16 + row) * KDIM;

    v8f acc[2][4];
    #pragma unroll
    for (int m = 0; m < 2; ++m)
        #pragma unroll
        for (int t = 0; t < 4; ++t)
            acc[m][t] = (v8f){0.f,0.f,0.f,0.f,0.f,0.f,0.f,0.f};

    for (int k0 = 0; k0 < KDIM; k0 += 32) {
        // prefetch next A tile lines into cache (global_prefetch_b8)
        if (k0 + 32 < KDIM) {
            __builtin_prefetch(arow0 + k0 + 32, 0, 3);
            __builtin_prefetch(arow1 + k0 + 32, 0, 3);
        }

        // A fragments: 16x32 bf16.  lanes 0-15: K = j (j<8), 16+j (j>=8)
        //              lanes 16-31: K = 8+j, 24+j   -> ksel*8 offset
        const v8bf lo0 = *(const v8bf*)(arow0 + k0 + ksel * 8);
        const v8bf hi0 = *(const v8bf*)(arow0 + k0 + 16 + ksel * 8);
        const v8bf lo1 = *(const v8bf*)(arow1 + k0 + ksel * 8);
        const v8bf hi1 = *(const v8bf*)(arow1 + k0 + 16 + ksel * 8);
        v16bf a0, a1;
        #pragma unroll
        for (int j = 0; j < 8; ++j) {
            a0[j] = lo0[j]; a0[8 + j] = hi0[j];
            a1[j] = lo1[j]; a1[8 + j] = hi1[j];
        }

        #pragma unroll
        for (int t = 0; t < 4; ++t) {
            // B fragment: 32x16 bf16. lane holds col = lane&15,
            // K = ksel*8 + j (j<8), 16 + ksel*8 + (j-8) (j>=8), same as A
            const int col = nwave + t * 16 + row;
            const v8bf blo = *(const v8bf*)(BT + (size_t)col * KDIM + k0 + ksel * 8);
            const v8bf bhi = *(const v8bf*)(BT + (size_t)col * KDIM + k0 + 16 + ksel * 8);
            v16bf b;
            #pragma unroll
            for (int j = 0; j < 8; ++j) { b[j] = blo[j]; b[8 + j] = bhi[j]; }
            acc[0][t] = __builtin_amdgcn_wmma_f32_16x16x32_bf16(
                            false, a0, false, b, (short)0, acc[0][t], false, false);
            acc[1][t] = __builtin_amdgcn_wmma_f32_16x16x32_bf16(
                            false, a1, false, b, (short)0, acc[1][t], false, false);
        }
    }

    // Store D: VGPR v -> M = v + 8*ksel ; N = lane&15
    #pragma unroll
    for (int m = 0; m < 2; ++m) {
        #pragma unroll
        for (int t = 0; t < 4; ++t) {
            const int col = nwave + t * 16 + row;
            #pragma unroll
            for (int v = 0; v < 8; ++v) {
                C[(size_t)(mbase + m * 16 + v + 8 * ksel) * DH + col] = acc[m][t][v];
            }
        }
    }
}

// ---------------------------------------------------------------------------
// Graph aggregation:  h[dst] += xw[src] * dinv[src]*dinv[dst]  (+ self loops)
// ---------------------------------------------------------------------------
__global__ void k_selfloop_init(const float* __restrict__ xw, const float* __restrict__ dinv,
                                float* __restrict__ h, int n) {
    size_t i = (size_t)blockIdx.x * blockDim.x + threadIdx.x;
    size_t total = (size_t)n * DH;
    size_t stride = (size_t)gridDim.x * blockDim.x;
    for (; i < total; i += stride) {
        float di = dinv[i >> 8];                 // DH == 256
        h[i] = xw[i] * di * di;
    }
}

__global__ __launch_bounds__(256)
void k_edge_agg(const int* __restrict__ src, const int* __restrict__ dst,
                const float* __restrict__ dinv, const float* __restrict__ xw,
                float* __restrict__ h, int e) {
    const int f = threadIdx.x;                   // one feature per thread (DH=256)
    for (int ed = blockIdx.x; ed < e; ed += gridDim.x) {
        const int s = src[ed];
        const int d = dst[ed];
        const float c = dinv[s] * dinv[d];
        atomicAdd(&h[(size_t)d * DH + f], xw[(size_t)s * DH + f] * c);
    }
}

// out32 / out16 may each be null; v = relu(h + bias)
__global__ void k_bias_relu(const float* __restrict__ h, const float* __restrict__ bias,
                            float* __restrict__ out32, bf16_t* __restrict__ out16, int n) {
    size_t i = (size_t)blockIdx.x * blockDim.x + threadIdx.x;
    size_t total = (size_t)n * DH;
    size_t stride = (size_t)gridDim.x * blockDim.x;
    for (; i < total; i += stride) {
        float v = h[i] + bias[i & (DH - 1)];
        v = v > 0.0f ? v : 0.0f;
        if (out32) out32[i] = v;
        if (out16) out16[i] = to_bf16(v);
    }
}

// ---------------------------------------------------------------------------
// Pooling + final FC
// ---------------------------------------------------------------------------
__global__ void k_zero(float* __restrict__ p, size_t n) {
    size_t i = (size_t)blockIdx.x * blockDim.x + threadIdx.x;
    size_t stride = (size_t)gridDim.x * blockDim.x;
    for (; i < n; i += stride) p[i] = 0.0f;
}

__global__ __launch_bounds__(256)
void k_pool(const float* __restrict__ h, const int* __restrict__ batch,
            float* __restrict__ sums, float* __restrict__ cnt, int n) {
    const int f = threadIdx.x;
    for (int i = blockIdx.x; i < n; i += gridDim.x) {
        const int g = batch[i];
        atomicAdd(&sums[(size_t)g * DH + f], h[(size_t)i * DH + f]);
        if (f == 0) atomicAdd(&cnt[g], 1.0f);
    }
}

__global__ __launch_bounds__(256)
void k_fc(const float* __restrict__ sums, const float* __restrict__ cnt,
          const float* __restrict__ wfc, const float* __restrict__ bfc,
          float* __restrict__ out) {
    __shared__ float red[256];
    const int g = blockIdx.x;
    const int f = threadIdx.x;
    float c = cnt[g];
    c = c < 1.0f ? 1.0f : c;
    red[f] = sums[(size_t)g * DH + f] / c * wfc[f];
    __syncthreads();
    #pragma unroll
    for (int s = 128; s > 0; s >>= 1) {
        if (f < s) red[f] += red[f + s];
        __syncthreads();
    }
    if (f == 0) out[g] = red[0] + bfc[0];
}

// ---------------------------------------------------------------------------
// Launch
// ---------------------------------------------------------------------------
extern "C" void kernel_launch(void* const* d_in, const int* in_sizes, int n_in,
                              void* d_out, int out_size, void* d_ws, size_t ws_size,
                              hipStream_t stream) {
    const float* x    = (const float*)d_in[0];   // [N, 512]
    const int*   ei   = (const int*)  d_in[1];   // [2, E]
    const int*   bat  = (const int*)  d_in[2];   // [N]
    const float* W1   = (const float*)d_in[3];   // [512, 256]
    const float* b1   = (const float*)d_in[4];   // [256]
    const float* W2   = (const float*)d_in[5];   // [256, 256]
    const float* b2   = (const float*)d_in[6];   // [256]
    const float* wfc  = (const float*)d_in[7];   // [256]
    const float* bfc  = (const float*)d_in[8];   // [1]
    float* out = (float*)d_out;                  // [128]

    const int* src = ei;
    const int* dst = ei + EE;

    // ---- workspace carve-up (256B aligned) ----
    char* ws = (char*)d_ws;
    size_t off = 0;
    auto carve = [&](size_t bytes) -> char* {
        char* p = ws + off;
        off += (bytes + 255) & ~(size_t)255;
        return p;
    };
    float*  dinv = (float*) carve((size_t)NN * 4);
    bf16_t* xh   = (bf16_t*)carve((size_t)NN * DIN * 2);
    bf16_t* w1t  = (bf16_t*)carve((size_t)DH * DIN * 2);   // [256][512]
    bf16_t* w2t  = (bf16_t*)carve((size_t)DH * DH  * 2);   // [256][256]
    float*  xw   = (float*) carve((size_t)NN * DH * 4);    // GEMM output (reused)
    float*  hbuf = (float*) carve((size_t)NN * DH * 4);    // aggregation buffer (reused)
    bf16_t* h1h  = (bf16_t*)carve((size_t)NN * DH * 2);    // bf16 hidden for GEMM2
    float*  sums = (float*) carve((size_t)GG * DH * 4);
    float*  cnt  = (float*) carve((size_t)GG * 4);
    (void)ws_size; (void)n_in; (void)in_sizes; (void)out_size;

    // ---- normalization coefficients ----
    k_init_deg     <<<256, 256, 0, stream>>>(dinv, NN);
    k_deg_edges    <<<1024, 256, 0, stream>>>(dst, dinv, EE);
    k_rsqrt_inplace<<<256, 256, 0, stream>>>(dinv, NN);

    // ---- bf16 conversions ----
    k_f32_to_bf16   <<<2048, 256, 0, stream>>>(x, xh, (size_t)NN * DIN);
    k_transpose_bf16<<<512, 256, 0, stream>>>(W1, w1t, DIN, DH);
    k_transpose_bf16<<<256, 256, 0, stream>>>(W2, w2t, DH, DH);

    // ---- layer 1: GEMM + aggregate + bias/relu (bf16 out) ----
    k_wmma_gemm<DIN><<<NN / 32, 128, 0, stream>>>(xh, w1t, xw, NN);
    k_selfloop_init  <<<2048, 256, 0, stream>>>(xw, dinv, hbuf, NN);
    k_edge_agg       <<<8192, 256, 0, stream>>>(src, dst, dinv, xw, hbuf, EE);
    k_bias_relu      <<<2048, 256, 0, stream>>>(hbuf, b1, (float*)nullptr, h1h, NN);

    // ---- layer 2: GEMM + aggregate + bias/relu (f32 out into xw) ----
    k_wmma_gemm<DH><<<NN / 32, 128, 0, stream>>>(h1h, w2t, xw, NN);
    k_selfloop_init  <<<2048, 256, 0, stream>>>(xw, dinv, hbuf, NN);
    k_edge_agg       <<<8192, 256, 0, stream>>>(src, dst, dinv, xw, hbuf, EE);
    k_bias_relu      <<<2048, 256, 0, stream>>>(hbuf, b2, xw, (bf16_t*)nullptr, NN);

    // ---- mean-pool + FC ----
    k_zero<<<64, 256, 0, stream>>>(sums, (size_t)(GG * DH));
    k_zero<<<1, 256, 0, stream>>>(cnt, (size_t)GG);
    k_pool<<<4096, 256, 0, stream>>>(xw, bat, sums, cnt, NN);
    k_fc  <<<GG, 256, 0, stream>>>(sums, cnt, wfc, bfc, out);
}
